// baseline_29154238005824
// MI455X (gfx1250) — matrix operand order repaired and checked by reading, not compiled, not run on hardware
//
#include <hip/hip_runtime.h>
#include <cstddef>

// ---------------------------------------------------------------------------
// Types for CDNA5 WMMA (wave32): v_wmma_f32_16x16x32_bf16
// ---------------------------------------------------------------------------
typedef __bf16 bf16_t;
typedef __attribute__((ext_vector_type(16))) __bf16 v16bf;
typedef __attribute__((ext_vector_type(8)))  __bf16 v8bf;
typedef __attribute__((ext_vector_type(4)))  __bf16 v4bf;
typedef __attribute__((ext_vector_type(8)))  float  v8f;

__device__ __forceinline__ bf16_t f2bf(float f) { return (bf16_t)f; }

__device__ __forceinline__ v4bf pack4(float4 v) {
  v4bf r;
  r[0] = (bf16_t)v.x; r[1] = (bf16_t)v.y; r[2] = (bf16_t)v.z; r[3] = (bf16_t)v.w;
  return r;
}

// ---------------------------------------------------------------------------
// A-fragment (16x32 bf16, M x K) from LDS.
// Layout (matches B): lanes 0-15 hold row M=lane, K=0..15;
// lanes 16-31 hold row M=lane-16, K=16..31.
// => per lane: 16 contiguous bf16 at k0+koff, koff=16*(lane>>4)
// ---------------------------------------------------------------------------
__device__ __forceinline__ v16bf load_a_lds(const bf16_t* act, int lda, int m0,
                                            int k0, int lane) {
  int row  = m0 + (lane & 15);
  int koff = k0 + ((lane >> 4) << 4);
  const bf16_t* base = act + row * lda + koff;
  v8bf lo = *(const v8bf*)(base);
  v8bf hi = *(const v8bf*)(base + 8);
  v16bf a;
#pragma unroll
  for (int i = 0; i < 8; ++i) { a[i] = lo[i]; a[i + 8] = hi[i]; }
  return a;
}

// ---------------------------------------------------------------------------
// B-fragment (32x16 bf16, K x N) from global bf16 weights, row-major [Nout,K]
// (PyTorch-style w[out][in] == B^T): lane n (0-15) holds column N=n, K=0..15;
// lane n+16 holds column N=n, K=16..31 -> 32 contiguous bytes per lane.
// ---------------------------------------------------------------------------
__device__ __forceinline__ v16bf load_b_gl(const bf16_t* __restrict__ w, int K,
                                           int n0, int k0, int lane) {
  int col = n0 + (lane & 15);
  int kh  = k0 + ((lane >> 4) << 4);
  const v8bf* q = (const v8bf*)(w + (size_t)col * K + kh);
  v8bf lo = q[0];
  v8bf hi = q[1];
  v16bf b;
#pragma unroll
  for (int i = 0; i < 8; ++i) { b[i] = lo[i]; b[i + 8] = hi[i]; }
  return b;
}

// ---------------------------------------------------------------------------
// One MLP layer on a 32-row activation tile resident in LDS.
// N-tile outer loop with all B fragments preloaded to registers, reused
// across both M-tiles (guaranteed weight-fragment reuse).
// C/D layout: VGPR r of lane l -> M = r + 8*(l>>4), N = l&15.
// ---------------------------------------------------------------------------
template <int KDIM, int NOUT>
__device__ __forceinline__ void mlp_layer(const bf16_t* act, int lda,
                                          const bf16_t* __restrict__ w,
                                          const float* __restrict__ bias,
                                          bf16_t* outbuf, int ldo,
                                          int lane, int wave) {
  constexpr int NT = NOUT / 16;   // N tiles
  constexpr int KT = KDIM / 32;   // K steps
  for (int nt = wave; nt < NT; nt += 8) {
    int n0 = nt * 16;
    v16bf bfrag[KT];
#pragma unroll
    for (int k = 0; k < KT; ++k) bfrag[k] = load_b_gl(w, KDIM, n0, k * 32, lane);
    int   ncol = lane & 15;
    float bv   = bias[n0 + ncol];
#pragma unroll
    for (int mt = 0; mt < 2; ++mt) {
      int m0 = mt * 16;
      v8f c = {};
#pragma unroll
      for (int k = 0; k < KT; ++k) {
        v16bf a = load_a_lds(act, lda, m0, k * 32, lane);
        c = __builtin_amdgcn_wmma_f32_16x16x32_bf16(false, a, false, bfrag[k],
                                                    (short)0, c, false, false);
      }
      int mbase = m0 + ((lane >> 4) << 3);
#pragma unroll
      for (int r = 0; r < 8; ++r) {
        float v = c[r] + bv;
        v = v > 0.f ? v : 0.f;
        outbuf[(mbase + r) * ldo + n0 + ncol] = f2bf(v);
      }
    }
  }
}

// ---------------------------------------------------------------------------
// Weight conversion: f32 -> bf16, stacking [lin_l | lin_r] into wnode
// [128 out, 256 in], and zero-padding w4 to [16 out, 64 in].
// ---------------------------------------------------------------------------
__global__ __launch_bounds__(256) void convert_weights_kernel(
    const float* __restrict__ ll, const float* __restrict__ lr,
    const float* __restrict__ w1, const float* __restrict__ w2,
    const float* __restrict__ w3, const float* __restrict__ w4,
    bf16_t* __restrict__ wnode, bf16_t* __restrict__ w1b,
    bf16_t* __restrict__ w2b, bf16_t* __restrict__ w3b,
    bf16_t* __restrict__ w4b) {
  int i = blockIdx.x * 256 + threadIdx.x;
  if (i < 32768) {  // wnode [128,256]
    int n = i >> 8, k = i & 255;
    wnode[i] = f2bf(k < 128 ? ll[n * 128 + k] : lr[n * 128 + (k - 128)]);
    return;
  }
  i -= 32768;
  if (i < 81920) { w1b[i] = f2bf(w1[i]); return; }  // [256,320]
  i -= 81920;
  if (i < 32768) { w2b[i] = f2bf(w2[i]); return; }  // [128,256]
  i -= 32768;
  if (i < 8192)  { w3b[i] = f2bf(w3[i]); return; }  // [64,128]
  i -= 8192;
  if (i < 1024) {                                    // [16(pad),64]
    int n = i >> 6, k = i & 63;
    w4b[i] = f2bf(n < 3 ? w4[n * 64 + k] : 0.f);
  }
}

__global__ __launch_bounds__(256) void zero_kernel(float* __restrict__ agg,
                                                   float* __restrict__ deg,
                                                   int N) {
  int i = blockIdx.x * 256 + threadIdx.x;
  if (i < N * 128) agg[i] = 0.f;
  if (i < N) deg[i] = 0.f;
}

// ---------------------------------------------------------------------------
// Edge scatter: agg[dst] += h[src] (f32 atomics), deg[dst] += 1.
// One thread per (edge, 4-float chunk): 32 threads per edge.
// ---------------------------------------------------------------------------
__global__ __launch_bounds__(256) void scatter_kernel(
    const float* __restrict__ hin, const int* __restrict__ ei,
    float* __restrict__ agg, float* __restrict__ deg, int E) {
  int t = blockIdx.x * 256 + threadIdx.x;
  int e = t >> 5, c = t & 31;
  if (e >= E) return;
  int s = ei[e];
  int d = ei[E + e];
  const float4 v = *(const float4*)(hin + (size_t)s * 128 + c * 4);
  float* dp = agg + (size_t)d * 128 + c * 4;
  atomicAdd(dp + 0, v.x);
  atomicAdd(dp + 1, v.y);
  atomicAdd(dp + 2, v.z);
  atomicAdd(dp + 3, v.w);
  if (c == 0) atomicAdd(deg + d, 1.0f);
}

// ---------------------------------------------------------------------------
// Node update: hout = relu([agg/max(deg,1) | hin] @ wnode^T + b), 32 nodes/WG.
// ---------------------------------------------------------------------------
__global__ __launch_bounds__(256) void node_kernel(
    const float* __restrict__ agg, const float* __restrict__ degv,
    const float* __restrict__ hin, const bf16_t* __restrict__ wnode,
    const float* __restrict__ bias, float* __restrict__ hout, int N) {
  constexpr int LDA = 264;  // 256 padded, multiple of 8
  __shared__ bf16_t bufA[32 * LDA];
  __shared__ float s_rdeg[32];
  int tid = threadIdx.x, lane = tid & 31, wave = tid >> 5;
  int n0blk = blockIdx.x * 32;
  bool full = (n0blk + 32 <= N);

  if (tid < 32) {
    int nd = n0blk + tid;
    float dg = (nd < N) ? degv[nd] : 1.f;
    s_rdeg[tid] = 1.f / (dg > 1.f ? dg : 1.f);
  }
  __syncthreads();

  // Stage [mean | x] as bf16, float4 loads + b64 LDS stores.
  const float4* agg4 = (const float4*)agg;
  const float4* hin4 = (const float4*)hin;
  for (int i = tid; i < 32 * 64; i += 256) {
    int m = i >> 6, c = i & 63;  // c indexes float4 chunks of the 256-wide row
    int node = n0blk + m;
    float4 v = make_float4(0.f, 0.f, 0.f, 0.f);
    if (node < N) {
      if (c < 32) {
        v = agg4[(size_t)node * 32 + c];
        float r = s_rdeg[m];
        v.x *= r; v.y *= r; v.z *= r; v.w *= r;
      } else {
        v = hin4[(size_t)node * 32 + (c - 32)];
      }
    }
    *(v4bf*)(bufA + m * LDA + c * 4) = pack4(v);
  }
  __syncthreads();

  // GEMM: 8 N-tiles, one per wave; B fragments fully register-resident.
  {
    int nn0 = wave * 16;
    v16bf bfrag[8];
#pragma unroll
    for (int k = 0; k < 8; ++k) bfrag[k] = load_b_gl(wnode, 256, nn0, k * 32, lane);
    int   ncol = lane & 15;
    float bv   = bias[nn0 + ncol];
#pragma unroll
    for (int mt = 0; mt < 2; ++mt) {
      int m0 = mt * 16;
      v8f c = {};
#pragma unroll
      for (int k = 0; k < 8; ++k) {
        v16bf a = load_a_lds(bufA, LDA, m0, k * 32, lane);
        c = __builtin_amdgcn_wmma_f32_16x16x32_bf16(false, a, false, bfrag[k],
                                                    (short)0, c, false, false);
      }
      int mbase = m0 + ((lane >> 4) << 3);
      if (full) {
#pragma unroll
        for (int r = 0; r < 8; ++r) {
          float v = c[r] + bv;
          hout[(size_t)(n0blk + mbase + r) * 128 + nn0 + ncol] = v > 0.f ? v : 0.f;
        }
      } else {
#pragma unroll
        for (int r = 0; r < 8; ++r) {
          int node = n0blk + mbase + r;
          if (node < N) {
            float v = c[r] + bv;
            hout[(size_t)node * 128 + nn0 + ncol] = v > 0.f ? v : 0.f;
          }
        }
      }
    }
  }
}

// ---------------------------------------------------------------------------
// Edge MLP: 32 edges per workgroup. Gather [h[src]|h[dst]|ef] (K=320) into
// LDS bf16, then 320->256->128->64->3 with WMMA, ping-ponging LDS buffers.
// ---------------------------------------------------------------------------
__global__ __launch_bounds__(256) void mlp_kernel(
    const float* __restrict__ h, const int* __restrict__ ei,
    const float* __restrict__ ef,
    const bf16_t* __restrict__ w1b, const bf16_t* __restrict__ w2b,
    const bf16_t* __restrict__ w3b, const bf16_t* __restrict__ w4b,
    const float* __restrict__ b1, const float* __restrict__ b2,
    const float* __restrict__ b3, const float* __restrict__ b4,
    float* __restrict__ out, int E) {
  constexpr int LD0 = 328;  // >=320, multiple of 8, padded
  constexpr int LD1 = 264;  // >=256, multiple of 8, padded
  __shared__ bf16_t buf0[32 * LD0];
  __shared__ bf16_t buf1[32 * LD1];
  __shared__ int s_src[32], s_dst[32];

  int tid = threadIdx.x, lane = tid & 31, wave = tid >> 5;
  int e0 = blockIdx.x * 32;
  bool full = (e0 + 32 <= E);

  if (tid < 32) {
    int e = e0 + tid;
    if (e >= E) e = E - 1;
    s_src[tid] = ei[e];
    s_dst[tid] = ei[E + e];
  }
  __syncthreads();

  // Stage pair features as bf16: [v_emb(128) | u_emb(128) | edge_feat(64)]
  // float4 global loads -> packed 4xbf16 LDS stores (80 float4 chunks / edge).
  const float4* h4  = (const float4*)h;
  const float4* ef4 = (const float4*)ef;
  for (int i = tid; i < 32 * 80; i += 256) {
    int e = i / 80;
    int c = i - e * 80;
    float4 v;
    if (c < 32) {
      v = h4[(size_t)s_src[e] * 32 + c];
    } else if (c < 64) {
      v = h4[(size_t)s_dst[e] * 32 + (c - 32)];
    } else {
      int ge = e0 + e;
      if (ge >= E) ge = E - 1;
      v = ef4[(size_t)ge * 16 + (c - 64)];
    }
    *(v4bf*)(buf0 + e * LD0 + c * 4) = pack4(v);
  }
  __syncthreads();

  mlp_layer<320, 256>(buf0, LD0, w1b, b1, buf1, LD1, lane, wave);
  __syncthreads();
  mlp_layer<256, 128>(buf1, LD1, w2b, b2, buf0, LD0, lane, wave);
  __syncthreads();
  mlp_layer<128, 64>(buf0, LD0, w3b, b3, buf1, LD1, lane, wave);
  __syncthreads();

  // Final layer: [32,64] @ [64,16(pad)] -> write 3 logits per edge.
  if (wave < 2) {
    int m0 = wave * 16;
    v8f c = {};
#pragma unroll
    for (int kk = 0; kk < 64; kk += 32) {
      v16bf a = load_a_lds(buf1, LD1, m0, kk, lane);
      v16bf b = load_b_gl(w4b, 64, 0, kk, lane);
      c = __builtin_amdgcn_wmma_f32_16x16x32_bf16(false, a, false, b,
                                                  (short)0, c, false, false);
    }
    int ncol  = lane & 15;
    int mbase = m0 + ((lane >> 4) << 3);
    if (ncol < 3) {
      float bv = b4[ncol];
      if (full) {
#pragma unroll
        for (int r = 0; r < 8; ++r)
          out[(size_t)(e0 + mbase + r) * 3 + ncol] = c[r] + bv;
      } else {
#pragma unroll
        for (int r = 0; r < 8; ++r) {
          int e = e0 + mbase + r;
          if (e < E) out[(size_t)e * 3 + ncol] = c[r] + bv;
        }
      }
    }
  }
}

// ---------------------------------------------------------------------------
// Host launch
// ---------------------------------------------------------------------------
extern "C" void kernel_launch(void* const* d_in, const int* in_sizes, int n_in,
                              void* d_out, int out_size, void* d_ws,
                              size_t ws_size, hipStream_t stream) {
  const float* x        = (const float*)d_in[0];
  const int*   ei       = (const int*)d_in[1];
  const float* ef       = (const float*)d_in[2];
  // d_in[3] = num_nodes scalar (unused; derived from sizes)
  const float* lin_l_w  = (const float*)d_in[4];
  const float* lin_l_b  = (const float*)d_in[5];
  const float* lin_r_w  = (const float*)d_in[6];
  const float* w1 = (const float*)d_in[7];  const float* b1 = (const float*)d_in[8];
  const float* w2 = (const float*)d_in[9];  const float* b2 = (const float*)d_in[10];
  const float* w3 = (const float*)d_in[11]; const float* b3 = (const float*)d_in[12];
  const float* w4 = (const float*)d_in[13]; const float* b4 = (const float*)d_in[14];
  float* out = (float*)d_out;

  const int N = in_sizes[0] / 128;  // nodes
  const int E = in_sizes[1] / 2;    // edges

  // Workspace carve-out
  size_t off = 0;
  char* ws = (char*)d_ws;
  auto carve = [&](size_t bytes) -> void* {
    void* p = ws + off;
    off += (bytes + 255) & ~(size_t)255;
    return p;
  };
  float*  agg   = (float*)carve((size_t)N * 128 * sizeof(float));
  float*  deg   = (float*)carve((size_t)N * sizeof(float));
  float*  h1    = (float*)carve((size_t)N * 128 * sizeof(float));
  float*  h2    = (float*)carve((size_t)N * 128 * sizeof(float));
  bf16_t* wnode = (bf16_t*)carve(128 * 256 * sizeof(bf16_t));
  bf16_t* w1b   = (bf16_t*)carve(256 * 320 * sizeof(bf16_t));
  bf16_t* w2b   = (bf16_t*)carve(128 * 256 * sizeof(bf16_t));
  bf16_t* w3b   = (bf16_t*)carve(64 * 128 * sizeof(bf16_t));
  bf16_t* w4b   = (bf16_t*)carve(16 * 64 * sizeof(bf16_t));

  // 1) Weights -> bf16 (156672 elements total)
  convert_weights_kernel<<<612, 256, 0, stream>>>(lin_l_w, lin_r_w, w1, w2, w3,
                                                  w4, wnode, w1b, w2b, w3b, w4b);

  // 2) Two SAGE layers (shared weights)
  const float* hin = x;
  float* houts[2] = {h1, h2};
  for (int l = 0; l < 2; ++l) {
    zero_kernel<<<(N * 128 + 255) / 256, 256, 0, stream>>>(agg, deg, N);
    scatter_kernel<<<(unsigned)(((size_t)E * 32 + 255) / 256), 256, 0, stream>>>(
        hin, ei, agg, deg, E);
    node_kernel<<<(N + 31) / 32, 256, 0, stream>>>(agg, deg, hin, wnode,
                                                   lin_l_b, houts[l], N);
    hin = houts[l];
  }

  // 3) Edge MLP (dominant GEMM work, bf16 WMMA)
  mlp_kernel<<<(E + 31) / 32, 256, 0, stream>>>(h2, ei, ef, w1b, w2b, w3b, w4b,
                                                b1, b2, b3, b4, out, E);
}
